// CausalSelfAttention_7971459301414
// MI455X (gfx1250) — compile-verified, matrix operand order checked by reading, not run on hardware
//
#include <hip/hip_runtime.h>

// CDNA5 / gfx1250, wave32. Causal self-attention, bf16 WMMA pipeline:
//   1-3) q/k/v = x @ W^T + b   (double-buffered bf16 WMMA GEMM; v transposed)
//   4)   flash attention (QK^T wmma -> online softmax -> P V wmma)
//   5)   out = y @ Wp^T + bp   (f32 output)

typedef __bf16 v16bf __attribute__((ext_vector_type(16)));
typedef float  v8f   __attribute__((ext_vector_type(8)));

#define B_ 4
#define T_ 2048
#define C_ 1024
#define H_ 16
#define D_ 64

__device__ __forceinline__ v8f wmma_bf16(v16bf a, v16bf b, v8f c) {
    // D = A(16x32 bf16) x B(32x16 bf16) + C(16x16 f32)
    return __builtin_amdgcn_wmma_f32_16x16x32_bf16(false, a, false, b,
                                                   (short)0, c, false, false);
}

// ---------------------------------------------------------------------------
// GEMM: Out = A @ W^T + bias.  A:(8192,1024) f32 or bf16, W:(1024,1024) f32.
// MODE 0: bf16 out, layout (B,H,T,D)   (q, k)
// MODE 1: bf16 out, layout (B,H,D,T)   (v transposed for PV wmma)
// MODE 2: f32  out, row-major (B*T, C) (final projection)
// Block: 128x128 tile, 256 threads = 8 waves (2 x 4), wave tile 64x32.
// Double-buffered LDS: one barrier per K-step; global->reg staging overlaps
// with the WMMA of the current slab.
// ---------------------------------------------------------------------------
template<int MODE, bool A_F32>
__global__ __launch_bounds__(256)
void gemm_bias_kernel(const void* __restrict__ Ap, const float* __restrict__ W,
                      const float* __restrict__ bias, void* __restrict__ Out)
{
    __shared__ alignas(32) __bf16 la[2][128 * 32];
    __shared__ alignas(32) __bf16 lb[2][128 * 32];

    const int tid  = threadIdx.x;
    const int lane = tid & 31;
    const int wid  = tid >> 5;
    const int ll   = lane & 15;     // fragment row/col within half
    const int lh   = lane >> 4;     // which 16-lane half
    const int wm0  = (wid >> 2) * 64;   // wave M origin in block: 0 or 64
    const int wn0  = (wid & 3) * 32;    // wave N origin in block: 0..96
    const int bm0  = blockIdx.y * 128;
    const int bn0  = blockIdx.x * 128;

    const v8f vz = {0.f, 0.f, 0.f, 0.f, 0.f, 0.f, 0.f, 0.f};
    v8f acc[4][2];
    #pragma unroll
    for (int i = 0; i < 4; i++)
        #pragma unroll
        for (int j = 0; j < 2; j++) acc[i][j] = vz;

    // staging: each of 256 threads moves 16 elements of A and of W per K-step
    const int sr = tid >> 1;         // 0..127 : tile row
    const int sc = (tid & 1) * 16;   // 0 or 16 : col offset within 32-wide slab

    const float*  Af = (const float*)Ap;
    const __bf16* Ab = (const __bf16*)Ap;

    float4 ra[4], rw[4];             // register staging (f32 path)
    v16bf  rab;                      // register staging (bf16 path)

    auto load_regs = [&](int k0) {
        if (A_F32) {
            #pragma unroll
            for (int u = 0; u < 4; u++)
                ra[u] = *(const float4*)&Af[(size_t)(bm0 + sr) * C_ + k0 + sc + u * 4];
        } else {
            rab = *(const v16bf*)&Ab[(size_t)(bm0 + sr) * C_ + k0 + sc];
        }
        #pragma unroll
        for (int u = 0; u < 4; u++)
            rw[u] = *(const float4*)&W[(size_t)(bn0 + sr) * C_ + k0 + sc + u * 4];
        // speculative prefetch two slabs ahead (L2-resident working set)
        __builtin_prefetch(&W[(size_t)(bn0 + sr) * C_ + k0 + sc + 64], 0, 1);
        if (A_F32)
            __builtin_prefetch(&Af[(size_t)(bm0 + sr) * C_ + k0 + sc + 64], 0, 1);
    };
    auto store_lds = [&](int bi) {
        if (A_F32) {
            #pragma unroll
            for (int u = 0; u < 4; u++) {
                la[bi][sr * 32 + sc + u * 4 + 0] = (__bf16)ra[u].x;
                la[bi][sr * 32 + sc + u * 4 + 1] = (__bf16)ra[u].y;
                la[bi][sr * 32 + sc + u * 4 + 2] = (__bf16)ra[u].z;
                la[bi][sr * 32 + sc + u * 4 + 3] = (__bf16)ra[u].w;
            }
        } else {
            *(v16bf*)&la[bi][sr * 32 + sc] = rab;
        }
        #pragma unroll
        for (int u = 0; u < 4; u++) {
            lb[bi][sr * 32 + sc + u * 4 + 0] = (__bf16)rw[u].x;
            lb[bi][sr * 32 + sc + u * 4 + 1] = (__bf16)rw[u].y;
            lb[bi][sr * 32 + sc + u * 4 + 2] = (__bf16)rw[u].z;
            lb[bi][sr * 32 + sc + u * 4 + 3] = (__bf16)rw[u].w;
        }
    };

    load_regs(0);
    store_lds(0);
    __syncthreads();

    const int NK = C_ / 32;
    for (int kt = 0; kt < NK; ++kt) {
        const int cur = kt & 1;
        if (kt + 1 < NK) load_regs((kt + 1) * 32);   // issue next slab early

        v16bf af[4], bf[2];
        #pragma unroll
        for (int i = 0; i < 4; i++)
            af[i] = *(const v16bf*)&la[cur][(wm0 + i * 16 + ll) * 32 + lh * 16];
        #pragma unroll
        for (int j = 0; j < 2; j++)
            bf[j] = *(const v16bf*)&lb[cur][(wn0 + j * 16 + ll) * 32 + lh * 16];

        #pragma unroll
        for (int i = 0; i < 4; i++)
            #pragma unroll
            for (int j = 0; j < 2; j++)
                acc[i][j] = wmma_bf16(af[i], bf[j], acc[i][j]);

        if (kt + 1 < NK) {
            store_lds(cur ^ 1);
            __syncthreads();   // one barrier per K-step
        }
    }

    // epilogue: C/D layout => element (m,n): m = p + 8*lh, n = ll (per 16x16)
    #pragma unroll
    for (int i = 0; i < 4; i++) {
        #pragma unroll
        for (int j = 0; j < 2; j++) {
            const int n  = bn0 + wn0 + j * 16 + ll;
            const float bv = bias[n];
            #pragma unroll
            for (int p = 0; p < 8; p++) {
                const int m = bm0 + wm0 + i * 16 + p + 8 * lh;
                const float val = acc[i][j][p] + bv;
                if (MODE == 2) {
                    ((float*)Out)[(size_t)m * C_ + n] = val;
                } else {
                    const int b = m / T_, t = m % T_;
                    const int h = n >> 6, dc = n & 63;
                    const size_t idx = (MODE == 0)
                        ? (((size_t)(b * H_ + h) * T_ + t) * D_ + dc)   // (B,H,T,D)
                        : (((size_t)(b * H_ + h) * D_ + dc) * T_ + t);  // (B,H,D,T)
                    ((__bf16*)Out)[idx] = (__bf16)val;
                }
            }
        }
    }
}

// ---------------------------------------------------------------------------
// Flash attention. Block = 128 threads = 4 waves; each wave owns a 16-row Q
// tile of one (b,h). Streams K/V in 32-column chunks, online softmax in f32.
// ---------------------------------------------------------------------------
__global__ __launch_bounds__(128)
void flash_attn_kernel(const __bf16* __restrict__ qh, const __bf16* __restrict__ kh,
                       const __bf16* __restrict__ vT, const float* __restrict__ mask,
                       __bf16* __restrict__ y)
{
    __shared__ alignas(32) __bf16 pbuf[4][16 * 32];  // wave-private P scratch

    const int tid  = threadIdx.x;
    const int lane = tid & 31;
    const int wid  = tid >> 5;
    const int ll   = lane & 15;
    const int lh   = lane >> 4;

    const int bh   = blockIdx.x >> 5;        // T/64 = 32 blocks per (b,h)
    const int tile = blockIdx.x & 31;
    const int b    = bh >> 4;                // H_ = 16
    const int h    = bh & 15;
    const int qr0  = tile * 64 + wid * 16;   // this wave's first query row

    const __bf16* qbase = qh + (size_t)bh * T_ * D_;
    const __bf16* kbase = kh + (size_t)bh * T_ * D_;
    const __bf16* vbase = vT + (size_t)bh * D_ * T_;
    const float*  mbase = mask + (size_t)b * T_;

    // Q fragments: 16x64 -> two 16x32 bf16 A-fragments
    v16bf qf[2];
    #pragma unroll
    for (int kk = 0; kk < 2; kk++)
        qf[kk] = *(const v16bf*)&qbase[(size_t)(qr0 + ll) * D_ + kk * 32 + lh * 16];

    const v8f vz = {0.f, 0.f, 0.f, 0.f, 0.f, 0.f, 0.f, 0.f};
    v8f o[4];
    #pragma unroll
    for (int j = 0; j < 4; j++) o[j] = vz;

    float rowM[8], rowL[8];
    #pragma unroll
    for (int p = 0; p < 8; p++) { rowM[p] = -1e30f; rowL[p] = 0.f; }

    const float scale = 0.125f;  // 1/sqrt(64)
    const int   sEnd  = qr0 + 16;

    for (int s0 = 0; s0 < sEnd; s0 += 32) {
        // prefetch next K/V chunk (L2-resident stream)
        if (s0 + 32 < sEnd) {
            __builtin_prefetch(&kbase[(size_t)(s0 + 32 + ll) * D_ + lh * 16], 0, 1);
            __builtin_prefetch(&vbase[(size_t)ll * T_ + s0 + 32 + lh * 16], 0, 1);
        }

        // ---- S = q @ k^T for columns [s0, s0+32) ----
        v8f s0v = vz, s1v = vz;
        #pragma unroll
        for (int kk = 0; kk < 2; kk++) {
            const v16bf k0f =
                *(const v16bf*)&kbase[(size_t)(s0 + ll) * D_ + kk * 32 + lh * 16];
            const v16bf k1f =
                *(const v16bf*)&kbase[(size_t)(s0 + 16 + ll) * D_ + kk * 32 + lh * 16];
            s0v = wmma_bf16(qf[kk], k0f, s0v);
            s1v = wmma_bf16(qf[kk], k1f, s1v);
        }

        // ---- scale, causal + padding mask, online softmax ----
        const int   kc0 = s0 + ll;
        const int   kc1 = s0 + 16 + ll;
        const float am0 = (1.f - mbase[kc0]) * -10000.f;
        const float am1 = (1.f - mbase[kc1]) * -10000.f;

        float p0[8], p1[8], alpha[8];
        #pragma unroll
        for (int p = 0; p < 8; p++) {
            const int qr = qr0 + p + 8 * lh;
            float a = (kc0 <= qr) ? s0v[p] * scale : -10000.f;  a += am0;
            float c = (kc1 <= qr) ? s1v[p] * scale : -10000.f;  c += am1;

            float mx = fmaxf(a, c);
            #pragma unroll
            for (int off = 1; off < 16; off <<= 1)      // stays inside 16-lane half
                mx = fmaxf(mx, __shfl_xor(mx, off, 32));
            const float nm = fmaxf(rowM[p], mx);

            a = __expf(a - nm);
            c = __expf(c - nm);
            p0[p] = a; p1[p] = c;

            float rs = a + c;
            #pragma unroll
            for (int off = 1; off < 16; off <<= 1)
                rs += __shfl_xor(rs, off, 32);

            alpha[p] = __expf(rowM[p] - nm);
            rowL[p]  = rowL[p] * alpha[p] + rs;
            rowM[p]  = nm;
        }

        #pragma unroll
        for (int j = 0; j < 4; j++)
            #pragma unroll
            for (int p = 0; p < 8; p++) o[j][p] *= alpha[p];

        // ---- P (accumulator layout) -> bf16 A-fragment layout via LDS ----
        __bf16* pb = pbuf[wid];
        #pragma unroll
        for (int p = 0; p < 8; p++) {
            pb[(p + 8 * lh) * 32 + ll]      = (__bf16)p0[p];
            pb[(p + 8 * lh) * 32 + 16 + ll] = (__bf16)p1[p];
        }
        // same-wave LDS ops are in-order; compiler inserts s_wait_dscnt
        const v16bf pf = *(const v16bf*)&pb[ll * 32 + lh * 16];

        // ---- O += P @ V   (V transposed: columns contiguous in T) ----
        #pragma unroll
        for (int j = 0; j < 4; j++) {
            const v16bf vf =
                *(const v16bf*)&vbase[(size_t)(j * 16 + ll) * T_ + s0 + lh * 16];
            o[j] = wmma_bf16(pf, vf, o[j]);
        }
    }

    // ---- normalize and store y (B,T,C) bf16 ----
    #pragma unroll
    for (int p = 0; p < 8; p++) {
        const float inv = 1.f / rowL[p];
        const int m = qr0 + p + 8 * lh;
        #pragma unroll
        for (int j = 0; j < 4; j++)
            y[((size_t)(b * T_ + m)) * C_ + h * D_ + j * 16 + ll] =
                (__bf16)(o[j][p] * inv);
    }
}

// ---------------------------------------------------------------------------
extern "C" void kernel_launch(void* const* d_in, const int* in_sizes, int n_in,
                              void* d_out, int out_size, void* d_ws, size_t ws_size,
                              hipStream_t stream) {
    (void)in_sizes; (void)n_in; (void)out_size; (void)ws_size;
    const float* x    = (const float*)d_in[0];
    const float* mask = (const float*)d_in[1];
    const float* Wq   = (const float*)d_in[2];
    const float* bq   = (const float*)d_in[3];
    const float* Wk   = (const float*)d_in[4];
    const float* bk   = (const float*)d_in[5];
    const float* Wv   = (const float*)d_in[6];
    const float* bv   = (const float*)d_in[7];
    const float* Wp   = (const float*)d_in[8];
    const float* bp   = (const float*)d_in[9];

    char* ws = (char*)d_ws;
    __bf16* qh = (__bf16*)(ws + (size_t)0);           // 16 MB: (B,H,T,D) bf16
    __bf16* kh = (__bf16*)(ws + ((size_t)16 << 20));  // 16 MB
    __bf16* vT = (__bf16*)(ws + ((size_t)32 << 20));  // 16 MB: (B,H,D,T) bf16
    __bf16* y  = (__bf16*)(ws + ((size_t)48 << 20));  // 16 MB: (B,T,C) bf16

    const dim3 gg(C_ / 128, (B_ * T_) / 128);  // (8, 64)
    gemm_bias_kernel<0, true ><<<gg, 256, 0, stream>>>(x, Wq, bq, qh);
    gemm_bias_kernel<0, true ><<<gg, 256, 0, stream>>>(x, Wk, bk, kh);
    gemm_bias_kernel<1, true ><<<gg, 256, 0, stream>>>(x, Wv, bv, vT);

    flash_attn_kernel<<<B_ * H_ * (T_ / 64), 128, 0, stream>>>(qh, kh, vT, mask, y);

    gemm_bias_kernel<2, false><<<gg, 256, 0, stream>>>(y, Wp, bp, d_out);
}
